// MultiHeadAttention_76656576299343
// MI455X (gfx1250) — matrix operand order repaired and checked by reading, not compiled, not run on hardware
//
#include <hip/hip_runtime.h>

// MHA for gfx1250: B=2, T=2048, C=1024, H=16, D=64.
// All matmuls via v_wmma_f32_16x16x32_bf16 (f32 accumulate).
// GEMM waves own 16x64 output tiles (4 independent WMMA accumulators).
// k-loops are software-pipelined AND unrolled by 2 with ping-pong fragment
// buffers, so prefetch needs no buffer-rotation register copies.
// Attention is flash-style: V loads issued early, K fragments ping-pong
// double-buffered, P re-layout through wave-private LDS.
// Workspace layout (40 MB total):
//   [0,8MB)   packed bf16 weights (Wq,Wk,Wv,Wo), fragment layout
//   [8,16MB)  Q   bf16 [B,H,T,D]
//   [16,24MB) K   bf16 [B,H,T,D]
//   [24,32MB) V^T bf16 [B,H,D,T]
//   [32,40MB) attention output bf16 [B,T,C]

typedef __attribute__((ext_vector_type(16))) __bf16 v16bf;
typedef __attribute__((ext_vector_type(8)))  __bf16 v8bf;
typedef __attribute__((ext_vector_type(8)))  float  v8f;
typedef __attribute__((ext_vector_type(4)))  float  v4f;

#define Csz   1024
#define Tsz   2048
#define Bsz   2
#define Hsz   16
#define Dsz   64
#define Msz   4096   // B*T

__device__ __forceinline__ v16bf cat8(v8bf lo, v8bf hi) {
  v16bf r;
#pragma unroll
  for (int i = 0; i < 8; ++i) { r[i] = lo[i]; r[8 + i] = hi[i]; }
  return r;
}

__device__ __forceinline__ v8f wmma_bf16(v16bf a, v16bf b, v8f c) {
  return __builtin_amdgcn_wmma_f32_16x16x32_bf16(false, a, false, b, (short)0, c,
                                                 false, false);
}

// A fragment (16x32 bf16) from an fp32 row: row = l, K = 16*hi + 0..15
__device__ __forceinline__ v16bf loadA_f32(const float* arow, int k0, int hi) {
  const v4f* ap4 = (const v4f*)(arow + k0 + 16 * hi);
  v4f x0 = ap4[0], x1 = ap4[1], x2 = ap4[2], x3 = ap4[3];
  v16bf a;
#pragma unroll
  for (int i = 0; i < 4; ++i) {
    a[i]      = (__bf16)x0[i];
    a[4 + i]  = (__bf16)x1[i];
    a[8 + i]  = (__bf16)x2[i];
    a[12 + i] = (__bf16)x3[i];
  }
  return a;
}

// A fragment from a bf16 row
__device__ __forceinline__ v16bf loadA_bf16(const __bf16* arow, int k0, int hi) {
  v8bf a0 = *(const v8bf*)(arow + k0 + 16 * hi);
  v8bf a1 = *(const v8bf*)(arow + k0 + 16 * hi + 8);
  return cat8(a0, a1);
}

// 4 packed-B fragments for k-step kt, column tiles ntile0..ntile0+3
__device__ __forceinline__ void loadB4(const __bf16* W, int kt, int ntile0,
                                       int lane, v16bf b[4]) {
  const __bf16* wbase = W + ((size_t)(kt * 64 + ntile0)) * 512 + lane * 16;
#pragma unroll
  for (int dc = 0; dc < 4; ++dc) b[dc] = *(const v16bf*)(wbase + dc * 512);
}

// ---------------------------------------------------------------------------
// Pack fp32 weights [K=1024, N=1024] (row-major, W[k][n]) into bf16 B-fragment
// layout: per (ktile of 32, ntile of 16) a 512-elem block; element index
// = lane*16 + e, where lane = (l | hi<<4), column n = ntile*16 + l,
// k = ktile*32 + hi*16 + e.
// ---------------------------------------------------------------------------
__global__ void __launch_bounds__(256) pack_w_kernel(
    const float* __restrict__ Wq, const float* __restrict__ Wk,
    const float* __restrict__ Wv, const float* __restrict__ Wo,
    __bf16* __restrict__ Wp) {
  int z = blockIdx.z;
  const float* W = (z == 0) ? Wq : (z == 1) ? Wk : (z == 2) ? Wv : Wo;
  size_t dst = (size_t)blockIdx.x * 256 + threadIdx.x;  // 0 .. 1M-1
  int e     = (int)(dst & 15);
  int lane  = (int)((dst >> 4) & 31);
  int blk   = (int)(dst >> 9);
  int ntile = blk & 63;
  int ktile = blk >> 6;
  int l  = lane & 15;
  int hi = lane >> 4;
  int n = ntile * 16 + l;
  int k = ktile * 32 + hi * 16 + e;
  Wp[(size_t)z * (Csz * Csz) + dst] = (__bf16)W[(size_t)k * Csz + n];
}

// ---------------------------------------------------------------------------
// QKV projection: wave computes 16x64 tile of X @ W + b; ping-pong pipelined.
// z = 0: Q -> [B,H,T,D]; z = 1: K -> [B,H,T,D]; z = 2: V -> [B,H,D,T] (transp.)
// grid: (256, 2, 3), block: 256 (8 waves).
// ---------------------------------------------------------------------------
__global__ void __launch_bounds__(256) gemm_qkv_kernel(
    const float* __restrict__ Xq, const float* __restrict__ Xk,
    const float* __restrict__ Xv, const __bf16* __restrict__ Wp,
    const float* __restrict__ bq, const float* __restrict__ bk,
    const float* __restrict__ bv, __bf16* __restrict__ Qb,
    __bf16* __restrict__ Kb, __bf16* __restrict__ Vtb) {
  int tid  = threadIdx.x;
  int lane = tid & 31, wave = tid >> 5;
  int l = lane & 15, hi = lane >> 4;
  int tileM  = blockIdx.x;                      // 0..255
  int ntile0 = (blockIdx.y * 8 + wave) * 4;     // 0..60, step 4
  int z = blockIdx.z;

  const float* X    = (z == 0) ? Xq : (z == 1) ? Xk : Xv;
  const __bf16* W   = Wp + (size_t)z * (Csz * Csz);
  const float* bias = (z == 0) ? bq : (z == 1) ? bk : bv;

  int row = tileM * 16 + l;
  const float* arow = X + (size_t)row * Csz;

  v8f acc[4] = {};
  v16bf a0 = loadA_f32(arow, 0, hi), a1;
  v16bf b0[4], b1[4];
  loadB4(W, 0, ntile0, lane, b0);

  // ping-pong: iteration consumes k-steps kt, kt+1; prefetches kt+1, kt+2
#pragma unroll 1
  for (int kt = 0; kt < Csz / 32 - 2; kt += 2) {
    a1 = loadA_f32(arow, (kt + 1) * 32, hi);
    loadB4(W, kt + 1, ntile0, lane, b1);
#pragma unroll
    for (int dc = 0; dc < 4; ++dc) acc[dc] = wmma_bf16(a0, b0[dc], acc[dc]);
    a0 = loadA_f32(arow, (kt + 2) * 32, hi);
    loadB4(W, kt + 2, ntile0, lane, b0);
#pragma unroll
    for (int dc = 0; dc < 4; ++dc) acc[dc] = wmma_bf16(a1, b1[dc], acc[dc]);
  }
  // epilogue: k-steps 30 (in set0) and 31
  a1 = loadA_f32(arow, (Csz / 32 - 1) * 32, hi);
  loadB4(W, Csz / 32 - 1, ntile0, lane, b1);
#pragma unroll
  for (int dc = 0; dc < 4; ++dc) acc[dc] = wmma_bf16(a0, b0[dc], acc[dc]);
#pragma unroll
  for (int dc = 0; dc < 4; ++dc) acc[dc] = wmma_bf16(a1, b1[dc], acc[dc]);

#pragma unroll
  for (int dc = 0; dc < 4; ++dc) {
    int n = (ntile0 + dc) * 16 + l;
    float bn = bias[n];
    int h = n >> 6, d = n & 63;
#pragma unroll
    for (int r = 0; r < 8; ++r) {
      int m  = tileM * 16 + r + 8 * hi;   // C-layout: row r (+8 upper half)
      int b_ = m >> 11, t = m & (Tsz - 1);
      float val = acc[dc][r] + bn;
      if (z == 2) {
        Vtb[(((size_t)(b_ * Hsz + h)) * Dsz + d) * Tsz + t] = (__bf16)val;
      } else {
        __bf16* dst = (z == 0) ? Qb : Kb;
        dst[(((size_t)(b_ * Hsz + h)) * Tsz + t) * Dsz + d] = (__bf16)val;
      }
    }
  }
}

// ---------------------------------------------------------------------------
// Flash attention: one wave owns a 16-row Q tile of one (b,h).
// Main loop handles 64 key columns (two 32-col halves) with ping-pong
// K-fragment buffers; V loads issue before the score WMMAs of each half.
// grid: 512 blocks, block 256 (8 waves).
// ---------------------------------------------------------------------------
__device__ __forceinline__ void load_kfrag(const __bf16* K, int j0, int l,
                                           int hi, v16bf kf[4]) {
  // kf[t2*2+half]: score tile t2 (cols j0+t2*16..), contraction d-half
#pragma unroll
  for (int t2 = 0; t2 < 2; ++t2)
#pragma unroll
    for (int half = 0; half < 2; ++half)
      kf[t2 * 2 + half] = *(const v16bf*)(K + (size_t)(j0 + t2 * 16 + l) * Dsz +
                                          half * 32 + hi * 16);
}

__device__ __forceinline__ void load_vfrag(const __bf16* Vt, int j0, int l,
                                           int hi, v16bf vf[4]) {
#pragma unroll
  for (int dc = 0; dc < 4; ++dc)
    vf[dc] = *(const v16bf*)(Vt + (size_t)(dc * 16 + l) * Tsz + j0 + hi * 16);
}

__global__ void __launch_bounds__(256) attn_kernel(
    const __bf16* __restrict__ Qb, const __bf16* __restrict__ Kb,
    const __bf16* __restrict__ Vtb, __bf16* __restrict__ Ob) {
  __shared__ __bf16 pbuf[8][16][32];  // per-wave private 16x32 P tile

  int tid  = threadIdx.x;
  int lane = tid & 31, w = tid >> 5;
  int l = lane & 15, hi = lane >> 4;
  int wgid  = blockIdx.x * 8 + w;      // 0..4095
  int qtile = wgid & 127;              // T/16
  int bh    = wgid >> 7;               // b*H + h

  const __bf16* Q  = Qb + ((size_t)bh * Tsz + qtile * 16) * Dsz;
  const __bf16* K  = Kb + (size_t)bh * Tsz * Dsz;
  const __bf16* Vt = Vtb + (size_t)bh * Dsz * Tsz;

  // Q A-fragments for d in [0,32) and [32,64)
  v16bf qf[2];
  const __bf16* qrow = Q + (size_t)l * Dsz;
#pragma unroll
  for (int half = 0; half < 2; ++half)
    qf[half] = loadA_bf16(qrow, half * 32, hi);

  v8f o0 = {}, o1 = {}, o2 = {}, o3 = {};
  float mrow[8], lrow[8];
#pragma unroll
  for (int r = 0; r < 8; ++r) { mrow[r] = -1e30f; lrow[r] = 0.0f; }

  const float scale = 0.125f;  // 1/sqrt(64)

  // softmax + P re-layout + P.V accumulation for one 32-column score pair
  auto process = [&](v8f c0, v8f c1, const v16bf vf[4]) {
#pragma unroll
    for (int r = 0; r < 8; ++r) {
      float s0 = c0[r] * scale, s1 = c1[r] * scale;
      float mx = fmaxf(s0, s1);
#pragma unroll
      for (int msk = 1; msk < 16; msk <<= 1)
        mx = fmaxf(mx, __shfl_xor(mx, msk, 16));
      float mnew = fmaxf(mrow[r], mx);
      float fac  = __expf(mrow[r] - mnew);
      float p0 = __expf(s0 - mnew), p1 = __expf(s1 - mnew);
      float rs = p0 + p1;
#pragma unroll
      for (int msk = 1; msk < 16; msk <<= 1)
        rs += __shfl_xor(rs, msk, 16);
      lrow[r] = lrow[r] * fac + rs;
      mrow[r] = mnew;
      o0[r] *= fac; o1[r] *= fac; o2[r] *= fac; o3[r] *= fac;
      pbuf[w][r + 8 * hi][l]      = (__bf16)p0;
      pbuf[w][r + 8 * hi][16 + l] = (__bf16)p1;
    }
    asm volatile("s_wait_dscnt 0" ::: "memory");  // wave-private LDS RAW
    v8bf plo = *(const v8bf*)(&pbuf[w][l][16 * hi]);
    v8bf phi = *(const v8bf*)(&pbuf[w][l][16 * hi + 8]);
    v16bf pf = cat8(plo, phi);
    o0 = wmma_bf16(pf, vf[0], o0);
    o1 = wmma_bf16(pf, vf[1], o1);
    o2 = wmma_bf16(pf, vf[2], o2);
    o3 = wmma_bf16(pf, vf[3], o3);
  };

  v16bf kfA[4], kfB[4];
  load_kfrag(K, 0, l, hi, kfA);

#pragma unroll 1
  for (int j0 = 0; j0 < Tsz; j0 += 64) {
    // ---- half 1: columns [j0, j0+32), fragments in kfA
    v16bf vf[4];
    load_vfrag(Vt, j0, l, hi, vf);
    v8f c0 = {}, c1 = {};
    c0 = wmma_bf16(qf[0], kfA[0], c0);
    c1 = wmma_bf16(qf[0], kfA[2], c1);
    c0 = wmma_bf16(qf[1], kfA[1], c0);
    c1 = wmma_bf16(qf[1], kfA[3], c1);
    load_kfrag(K, j0 + 32, l, hi, kfB);       // prefetch, hidden by softmax
    process(c0, c1, vf);

    // ---- half 2: columns [j0+32, j0+64), fragments in kfB
    v16bf vg[4];
    load_vfrag(Vt, j0 + 32, l, hi, vg);
    v8f d0 = {}, d1 = {};
    d0 = wmma_bf16(qf[0], kfB[0], d0);
    d1 = wmma_bf16(qf[0], kfB[2], d1);
    d0 = wmma_bf16(qf[1], kfB[1], d0);
    d1 = wmma_bf16(qf[1], kfB[3], d1);
    load_kfrag(K, (j0 + 64) & (Tsz - 1), l, hi, kfA);  // wraps on last iter
    process(d0, d1, vg);
  }

  // ---- epilogue: divide by row sums, store to [B,T,C] bf16
  int h = bh & 15, b = bh >> 4;
#pragma unroll
  for (int r = 0; r < 8; ++r) {
    int t = qtile * 16 + r + 8 * hi;
    float inv = 1.0f / lrow[r];
    size_t base = ((size_t)b * Tsz + t) * Csz + h * Dsz;
    Ob[base + 0 * 16 + l] = (__bf16)(o0[r] * inv);
    Ob[base + 1 * 16 + l] = (__bf16)(o1[r] * inv);
    Ob[base + 2 * 16 + l] = (__bf16)(o2[r] * inv);
    Ob[base + 3 * 16 + l] = (__bf16)(o3[r] * inv);
  }
}

// ---------------------------------------------------------------------------
// Output projection: bf16 A [4096,1024] @ packed Wo + bo -> fp32 out.
// Wave computes 16x64 tile; ping-pong pipelined.  grid: (256, 2), block 256.
// ---------------------------------------------------------------------------
__global__ void __launch_bounds__(256) gemm_out_kernel(
    const __bf16* __restrict__ A, const __bf16* __restrict__ Wp,
    const float* __restrict__ bias, float* __restrict__ out) {
  int tid  = threadIdx.x;
  int lane = tid & 31, wave = tid >> 5;
  int l = lane & 15, hi = lane >> 4;
  int tileM  = blockIdx.x;
  int ntile0 = (blockIdx.y * 8 + wave) * 4;

  int row = tileM * 16 + l;
  const __bf16* arow = A + (size_t)row * Csz;

  v8f acc[4] = {};
  v16bf a0 = loadA_bf16(arow, 0, hi), a1;
  v16bf b0[4], b1[4];
  loadB4(Wp, 0, ntile0, lane, b0);

#pragma unroll 1
  for (int kt = 0; kt < Csz / 32 - 2; kt += 2) {
    a1 = loadA_bf16(arow, (kt + 1) * 32, hi);
    loadB4(Wp, kt + 1, ntile0, lane, b1);
#pragma unroll
    for (int dc = 0; dc < 4; ++dc) acc[dc] = wmma_bf16(a0, b0[dc], acc[dc]);
    a0 = loadA_bf16(arow, (kt + 2) * 32, hi);
    loadB4(Wp, kt + 2, ntile0, lane, b0);
#pragma unroll
    for (int dc = 0; dc < 4; ++dc) acc[dc] = wmma_bf16(a1, b1[dc], acc[dc]);
  }
  a1 = loadA_bf16(arow, (Csz / 32 - 1) * 32, hi);
  loadB4(Wp, Csz / 32 - 1, ntile0, lane, b1);
#pragma unroll
  for (int dc = 0; dc < 4; ++dc) acc[dc] = wmma_bf16(a0, b0[dc], acc[dc]);
#pragma unroll
  for (int dc = 0; dc < 4; ++dc) acc[dc] = wmma_bf16(a1, b1[dc], acc[dc]);

#pragma unroll
  for (int dc = 0; dc < 4; ++dc) {
    int n = (ntile0 + dc) * 16 + l;
    float bn = bias[n];
#pragma unroll
    for (int r = 0; r < 8; ++r) {
      int m = tileM * 16 + r + 8 * hi;
      out[(size_t)m * Csz + n] = acc[dc][r] + bn;
    }
  }
}

// ---------------------------------------------------------------------------
extern "C" void kernel_launch(void* const* d_in, const int* in_sizes, int n_in,
                              void* d_out, int out_size, void* d_ws,
                              size_t ws_size, hipStream_t stream) {
  const float* query = (const float*)d_in[0];
  const float* key_  = (const float*)d_in[1];
  const float* value = (const float*)d_in[2];
  const float* Wq = (const float*)d_in[3];
  const float* bq = (const float*)d_in[4];
  const float* Wk = (const float*)d_in[5];
  const float* bk = (const float*)d_in[6];
  const float* Wv = (const float*)d_in[7];
  const float* bv = (const float*)d_in[8];
  const float* Wo = (const float*)d_in[9];
  const float* bo = (const float*)d_in[10];
  float* out = (float*)d_out;

  char* ws = (char*)d_ws;
  __bf16* Wp  = (__bf16*)(ws);                        //  8 MB: 4 packed weights
  __bf16* Qb  = (__bf16*)(ws + (8ull << 20));         //  8 MB
  __bf16* Kb  = (__bf16*)(ws + (16ull << 20));        //  8 MB
  __bf16* Vtb = (__bf16*)(ws + (24ull << 20));        //  8 MB
  __bf16* Ob  = (__bf16*)(ws + (32ull << 20));        //  8 MB

  pack_w_kernel<<<dim3(Csz * Csz / 256, 1, 4), 256, 0, stream>>>(Wq, Wk, Wv, Wo,
                                                                 Wp);
  gemm_qkv_kernel<<<dim3(Msz / 16, 2, 3), 256, 0, stream>>>(
      query, key_, value, Wp, bq, bk, bv, Qb, Kb, Vtb);
  attn_kernel<<<dim3((Bsz * Hsz * (Tsz / 16)) / 8), 256, 0, stream>>>(Qb, Kb,
                                                                      Vtb, Ob);
  gemm_out_kernel<<<dim3(Msz / 16, 2), 256, 0, stream>>>(
      Ob, Wp + 3ull * (Csz * Csz), bo, out);
}
